// LiDARToBEV_86363202388226
// MI455X (gfx1250) — matrix operand order repaired and checked by reading, not compiled, not run on hardware
//
#include <hip/hip_runtime.h>
#include <hip/hip_bf16.h>

// ---------------------------------------------------------------------------
// LiDAR -> BEV pipeline for MI455X (gfx1250, wave32, WMMA)
//   voxelize -> conv1(1->32) -> conv2(32->64, WMMA) -> conv3(64->64, WMMA)
//   -> fused max-over-D via global_atomic_max_num_f32 into d_out.
// Activations kept channels-last bf16 in halo-padded (66^3) buffers so the
// implicit-GEMM K dimension (tap-major, channel-minor) is contiguous.
// K loops are fully unrolled so every A/B address folds into an immediate
// offset on global_load_b128 off a single per-lane base pointer.
// ---------------------------------------------------------------------------

typedef __bf16 bf16_t;
typedef __bf16 v16bf __attribute__((ext_vector_type(16)));
typedef __bf16 v8bf  __attribute__((ext_vector_type(8)));
typedef float  v8f   __attribute__((ext_vector_type(8)));

#define GD    64
#define GP    66     // padded (1-voxel halo)
#define BATCH 2
#define BN_EPS 1e-5f

// ---------------------------------------------------------------- zero fill
__global__ void zero_u128_kernel(uint4* __restrict__ p, long long n4) {
  long long i = (long long)blockIdx.x * blockDim.x + threadIdx.x;
  long long stride = (long long)gridDim.x * blockDim.x;
  uint4 z; z.x = 0u; z.y = 0u; z.z = 0u; z.w = 0u;
  for (; i < n4; i += stride) p[i] = z;
}

// ------------------------------------------------- fold BN into scale/bias
// params layout: [0,32) s1 | [32,64) b1 | [64,128) s2 | [128,192) b2
//                [192,256) s3 | [256,320) b3
__global__ void prep_params_kernel(
    const float* b1, const float* g1, const float* be1, const float* m1, const float* v1,
    const float* b2, const float* g2, const float* be2, const float* m2, const float* v2,
    const float* b3, const float* g3, const float* be3, const float* m3, const float* v3,
    float* __restrict__ params) {
  int c = threadIdx.x;
  if (c < 32) {
    float inv = g1[c] * rsqrtf(v1[c] + BN_EPS);
    params[c]      = inv;
    params[32 + c] = fmaf(b1[c], inv, be1[c] - m1[c] * inv);
  } else if (c < 96) {
    int k = c - 32;
    float inv = g2[k] * rsqrtf(v2[k] + BN_EPS);
    params[64 + k]  = inv;
    params[128 + k] = fmaf(b2[k], inv, be2[k] - m2[k] * inv);
  } else if (c < 160) {
    int k = c - 96;
    float inv = g3[k] * rsqrtf(v3[k] + BN_EPS);
    params[192 + k] = inv;
    params[256 + k] = fmaf(b3[k], inv, be3[k] - m3[k] * inv);
  }
}

// ------------------------------------------------- pack weights -> B frags
// B-fragment layout (v_wmma 16x16x32 bf16, 32x16 B matrix, wave32):
//   lane n (0..15):  N = n,  K = {0..7, 16..23}
//   lane n+16:       N = n,  K = {8..15, 24..31}
// dst[((kb*4 + ctile)*32 + lane)*16 + idx] -> 32 contiguous bytes per lane.
__global__ void pack_w_kernel(const float* __restrict__ w, bf16_t* __restrict__ dst,
                              int Cin, int total) {
  int t = blockIdx.x * blockDim.x + threadIdx.x;
  if (t >= total) return;
  int kb   = t >> 11;
  int r    = t & 2047;
  int ct   = r >> 9;
  int r2   = r & 511;
  int lane = r2 >> 4;
  int idx  = r2 & 15;
  int cout = ct * 16 + (lane & 15);
  int K    = ((idx & 8) << 1) + ((lane >> 4) & 1) * 8 + (idx & 7);
  int kk   = kb * 32 + K;              // K ordering: tap*Cin + c
  int tap  = kk / Cin;
  int c    = kk - tap * Cin;
  float v  = w[((size_t)cout * Cin + c) * 27 + tap];
  dst[t] = (bf16_t)v;
}

// ---------------------------------------------------------- min reduction
__global__ void rmin_kernel(const float* __restrict__ pc, float* __restrict__ minb, int N) {
  __shared__ float s[256];
  int row = blockIdx.x;                       // row = b*3 + coord
  const float* p = pc + (size_t)row * N;
  float m = 3.4e38f;
  for (int i = threadIdx.x; i < N; i += 256) m = fminf(m, p[i]);
  s[threadIdx.x] = m;
  __syncthreads();
  for (int off = 128; off > 0; off >>= 1) {
    if (threadIdx.x < off) s[threadIdx.x] = fminf(s[threadIdx.x], s[threadIdx.x + off]);
    __syncthreads();
  }
  if (threadIdx.x == 0) minb[row] = s[0];
}

// --------------------------------------------------------------- voxelize
__global__ void voxelize_kernel(const float* __restrict__ pc, const float* __restrict__ minb,
                                float* __restrict__ grid, int N) {
  int idx = blockIdx.x * blockDim.x + threadIdx.x;
  if (idx >= BATCH * N) return;
  int b = idx / N;
  int n = idx - b * N;
  const float* p = pc + (size_t)b * 3 * N;
  float x = p[n], y = p[N + n], z = p[2 * N + n];
  int gx = min(max((int)floorf((x - minb[b * 3 + 0]) * 10.0f), 0), 63);
  int gy = min(max((int)floorf((y - minb[b * 3 + 1]) * 10.0f), 0), 63);
  int gz = min(max((int)floorf((z - minb[b * 3 + 2]) * 10.0f), 0), 63);
  grid[(((size_t)b * GD + gx) * GD + gy) * GD + gz] = 1.0f;  // idempotent
}

// ------------------------------------------------- conv1 (1->32), direct
// Occupancy is one-hot: conv = sum of weights over occupied neighbors.
__global__ void conv1_kernel(const float* __restrict__ grid, const float* __restrict__ w1,
                             const float* __restrict__ params, bf16_t* __restrict__ act1) {
  int idx = blockIdx.x * blockDim.x + threadIdx.x;   // BATCH*64^3 exact
  int w = idx & 63;
  int h = (idx >> 6) & 63;
  int d = (idx >> 12) & 63;
  int b = idx >> 18;
  float acc[32];
#pragma unroll
  for (int c = 0; c < 32; ++c) acc[c] = 0.f;
  for (int dz = 0; dz < 3; ++dz) {
    int z = d + dz - 1; if ((unsigned)z >= (unsigned)GD) continue;
    for (int dy = 0; dy < 3; ++dy) {
      int yy = h + dy - 1; if ((unsigned)yy >= (unsigned)GD) continue;
      for (int dx = 0; dx < 3; ++dx) {
        int xx = w + dx - 1; if ((unsigned)xx >= (unsigned)GD) continue;
        float occ = grid[(((size_t)b * GD + z) * GD + yy) * GD + xx];
        if (occ != 0.f) {
          int t = (dz * 3 + dy) * 3 + dx;
#pragma unroll
          for (int c = 0; c < 32; ++c) acc[c] += w1[c * 27 + t];
        }
      }
    }
  }
  const float* scale = params;         // s1
  const float* bias  = params + 32;    // b1 (conv bias folded with BN)
  size_t obase = ((((size_t)(b * GP + d + 1)) * GP + (h + 1)) * GP + (w + 1)) * 32;
#pragma unroll
  for (int c = 0; c < 32; ++c) {
    float yv = fmaxf(fmaf(acc[c], scale[c], bias[c]), 0.f);
    act1[obase + c] = (bf16_t)yv;
  }
}

// ------------------------------------------------- conv2 (32->64), WMMA
// Block = (b,d,h). 8 waves: wave w -> W-tile (w&3), Cout tiles {2*(w>>2), +1}.
// A-fragment per ISA 16-bit A layout: lane<16 K{0..7,16..23}, lane>=16 K{8..15,24..31}.
// Fully unrolled: all A/B addresses are immediate offsets off abase/bbase.
__global__ void __launch_bounds__(256) conv2_kernel(const bf16_t* __restrict__ act1,
                                                    const bf16_t* __restrict__ wp,
                                                    const float* __restrict__ params,
                                                    bf16_t* __restrict__ act2) {
  int blk = blockIdx.x;
  int h = blk & 63;
  int d = (blk >> 6) & 63;
  int b = blk >> 12;
  int lane = threadIdx.x & 31;
  int wave = threadIdx.x >> 5;
  int w0   = (wave & 3) * 16;
  int ct0  = (wave >> 2) * 2;
  int mrow = lane & 15;
  int c0   = (lane & 16) ? 8 : 0;

  const bf16_t* abase = act1 +
      ((((size_t)(b * GP + d)) * GP + h) * GP + (w0 + mrow)) * 32 + c0;
  const bf16_t* bbase = wp + ((size_t)(ct0 * 32 + lane)) * 16;
  __builtin_prefetch(bbase, 0, 3);
  __builtin_prefetch(bbase + 16384, 0, 3);

  v8f acc0 = {0, 0, 0, 0, 0, 0, 0, 0};
  v8f acc1 = {0, 0, 0, 0, 0, 0, 0, 0};

#pragma unroll
  for (int t = 0; t < 27; ++t) {
    const int dz = t / 9;
    const int dy = (t / 3) % 3;
    const int dx = t % 3;
    const bf16_t* ap = abase + ((dz * GP + dy) * GP + dx) * 32;   // const offset
    v8bf lo = *(const v8bf*)ap;
    v8bf hi = *(const v8bf*)(ap + 16);
    v16bf A = __builtin_shufflevector(lo, hi, 0, 1, 2, 3, 4, 5, 6, 7,
                                      8, 9, 10, 11, 12, 13, 14, 15);
    const bf16_t* bp = bbase + t * 2048;                          // const offset
    v16bf B0 = *(const v16bf*)bp;
    v16bf B1 = *(const v16bf*)(bp + 512);
    acc0 = __builtin_amdgcn_wmma_f32_16x16x32_bf16(false, A, false, B0,
                                                   (short)0, acc0, false, false);
    acc1 = __builtin_amdgcn_wmma_f32_16x16x32_bf16(false, A, false, B1,
                                                   (short)0, acc1, false, false);
  }

  const float* scale = params + 64;
  const float* bias  = params + 128;
  int Mbase = (lane & 16) ? 8 : 0;
  int cA = ct0 * 16 + mrow;
  int cB = cA + 16;
  float sA = scale[cA], bA = bias[cA];
  float sB = scale[cB], bB = bias[cB];
#pragma unroll
  for (int v = 0; v < 8; ++v) {
    int wpix = w0 + Mbase + v;
    size_t obase = ((((size_t)(b * GP + d + 1)) * GP + (h + 1)) * GP + (wpix + 1)) * 64;
    float y0 = fmaxf(fmaf(acc0[v], sA, bA), 0.f);
    float y1 = fmaxf(fmaf(acc1[v], sB, bB), 0.f);
    act2[obase + cA] = (bf16_t)y0;
    act2[obase + cB] = (bf16_t)y1;
  }
}

// ------------------------------------- conv3 (64->64), WMMA + fused max-D
__global__ void __launch_bounds__(256) conv3_kernel(const bf16_t* __restrict__ act2,
                                                    const bf16_t* __restrict__ wp,
                                                    const float* __restrict__ params,
                                                    float* __restrict__ out) {
  int blk = blockIdx.x;
  int h = blk & 63;
  int d = (blk >> 6) & 63;
  int b = blk >> 12;
  int lane = threadIdx.x & 31;
  int wave = threadIdx.x >> 5;
  int w0   = (wave & 3) * 16;
  int ct0  = (wave >> 2) * 2;
  int mrow = lane & 15;
  int c0   = (lane & 16) ? 8 : 0;

  const bf16_t* abase = act2 +
      ((((size_t)(b * GP + d)) * GP + h) * GP + (w0 + mrow)) * 64 + c0;
  const bf16_t* bbase = wp + ((size_t)(ct0 * 32 + lane)) * 16;
  __builtin_prefetch(bbase, 0, 3);
  __builtin_prefetch(bbase + 32768, 0, 3);

  v8f acc0 = {0, 0, 0, 0, 0, 0, 0, 0};
  v8f acc1 = {0, 0, 0, 0, 0, 0, 0, 0};

#pragma unroll
  for (int kb = 0; kb < 54; ++kb) {          // 27 taps x 2 K-blocks (Cin=64)
    const int t  = kb >> 1;
    const int cb = (kb & 1) * 32;
    const int dz = t / 9;
    const int dy = (t / 3) % 3;
    const int dx = t % 3;
    const bf16_t* ap = abase + ((dz * GP + dy) * GP + dx) * 64 + cb;  // const offset
    v8bf lo = *(const v8bf*)ap;
    v8bf hi = *(const v8bf*)(ap + 16);
    v16bf A = __builtin_shufflevector(lo, hi, 0, 1, 2, 3, 4, 5, 6, 7,
                                      8, 9, 10, 11, 12, 13, 14, 15);
    const bf16_t* bp = bbase + kb * 2048;                             // const offset
    v16bf B0 = *(const v16bf*)bp;
    v16bf B1 = *(const v16bf*)(bp + 512);
    acc0 = __builtin_amdgcn_wmma_f32_16x16x32_bf16(false, A, false, B0,
                                                   (short)0, acc0, false, false);
    acc1 = __builtin_amdgcn_wmma_f32_16x16x32_bf16(false, A, false, B1,
                                                   (short)0, acc1, false, false);
  }

  const float* scale = params + 192;
  const float* bias  = params + 256;
  int Mbase = (lane & 16) ? 8 : 0;
  int cA = ct0 * 16 + mrow;
  int cB = cA + 16;
  float sA = scale[cA], bA = bias[cA];
  float sB = scale[cB], bB = bias[cB];
#pragma unroll
  for (int v = 0; v < 8; ++v) {
    int wpix = w0 + Mbase + v;
    float y0 = fmaxf(fmaf(acc0[v], sA, bA), 0.f);
    float y1 = fmaxf(fmaf(acc1[v], sB, bB), 0.f);
    // out[b, c, h, w] = max over d; out zero-initialized, ReLU >= 0.
    const float* p0 = out + ((((size_t)b * 64 + cA) * 64 + h) * 64 + wpix);
    const float* p1 = out + ((((size_t)b * 64 + cB) * 64 + h) * 64 + wpix);
    asm volatile("global_atomic_max_num_f32 %0, %1, off"
                 :: "v"((unsigned long long)p0), "v"(y0) : "memory");
    asm volatile("global_atomic_max_num_f32 %0, %1, off"
                 :: "v"((unsigned long long)p1), "v"(y1) : "memory");
  }
}

// ---------------------------------------------------------------------------
extern "C" void kernel_launch(void* const* d_in, const int* in_sizes, int n_in,
                              void* d_out, int out_size, void* d_ws, size_t ws_size,
                              hipStream_t stream) {
  const float* pc  = (const float*)d_in[0];
  const float* w1  = (const float*)d_in[1];
  const float* b1  = (const float*)d_in[2];
  const float* g1  = (const float*)d_in[3];
  const float* be1 = (const float*)d_in[4];
  const float* m1  = (const float*)d_in[5];
  const float* v1  = (const float*)d_in[6];
  const float* w2  = (const float*)d_in[7];
  const float* b2  = (const float*)d_in[8];
  const float* g2  = (const float*)d_in[9];
  const float* be2 = (const float*)d_in[10];
  const float* m2  = (const float*)d_in[11];
  const float* v2  = (const float*)d_in[12];
  const float* w3  = (const float*)d_in[13];
  const float* b3  = (const float*)d_in[14];
  const float* g3  = (const float*)d_in[15];
  const float* be3 = (const float*)d_in[16];
  const float* m3  = (const float*)d_in[17];
  const float* v3  = (const float*)d_in[18];

  const int N = in_sizes[0] / (BATCH * 3);     // points per cloud

  // ---- workspace layout (~113 MB total, 256B-aligned slots) ----
  char* ws = (char*)d_ws;
  size_t o = 0;
  auto alloc = [&](size_t bytes) {
    size_t r = o;
    o += (bytes + 255) & ~(size_t)255;
    return r;
  };
  size_t off_minb   = alloc(6 * sizeof(float));
  size_t off_params = alloc(320 * sizeof(float));
  size_t off_wp2    = alloc((size_t)27 * 2048 * sizeof(bf16_t));  // 27 kb x 4 ct x 32 x 16
  size_t off_wp3    = alloc((size_t)54 * 2048 * sizeof(bf16_t));
  size_t off_grid   = alloc((size_t)BATCH * GD * GD * GD * sizeof(float));
  size_t off_act1   = alloc((size_t)BATCH * GP * GP * GP * 32 * sizeof(bf16_t));
  size_t off_act2   = alloc((size_t)BATCH * GP * GP * GP * 64 * sizeof(bf16_t));
  size_t zero_begin = off_grid;
  size_t zero_bytes = o - off_grid;            // grid + both padded activations
  (void)ws_size; (void)n_in;

  float*  minb   = (float*)(ws + off_minb);
  float*  params = (float*)(ws + off_params);
  bf16_t* wp2    = (bf16_t*)(ws + off_wp2);
  bf16_t* wp3    = (bf16_t*)(ws + off_wp3);
  float*  grid   = (float*)(ws + off_grid);
  bf16_t* act1   = (bf16_t*)(ws + off_act1);
  bf16_t* act2   = (bf16_t*)(ws + off_act2);
  float*  out    = (float*)d_out;

  // 1) zero halo'd activations + occupancy grid + output accumulator
  {
    long long n4 = (long long)(zero_bytes / 16);
    zero_u128_kernel<<<8192, 256, 0, stream>>>((uint4*)(ws + zero_begin), n4);
    long long n4o = (long long)out_size / 4;
    zero_u128_kernel<<<512, 256, 0, stream>>>((uint4*)out, n4o);
  }

  // 2) fold BN into scale/bias
  prep_params_kernel<<<1, 160, 0, stream>>>(b1, g1, be1, m1, v1,
                                            b2, g2, be2, m2, v2,
                                            b3, g3, be3, m3, v3, params);

  // 3) pack conv2/conv3 weights into WMMA B-fragment layout
  {
    int tot2 = 27 * 2048;
    int tot3 = 54 * 2048;
    pack_w_kernel<<<(tot2 + 255) / 256, 256, 0, stream>>>(w2, wp2, 32, tot2);
    pack_w_kernel<<<(tot3 + 255) / 256, 256, 0, stream>>>(w3, wp3, 64, tot3);
  }

  // 4) per-(batch,coord) min, then voxelize
  rmin_kernel<<<BATCH * 3, 256, 0, stream>>>(pc, minb, N);
  voxelize_kernel<<<(BATCH * N + 255) / 256, 256, 0, stream>>>(pc, minb, grid, N);

  // 5) conv1 direct (occupancy one-hot)
  conv1_kernel<<<(BATCH * GD * GD * GD) / 256, 256, 0, stream>>>(grid, w1, params, act1);

  // 6) conv2 / conv3 implicit-GEMM WMMA; conv3 fuses max-over-D via atomics
  conv2_kernel<<<BATCH * GD * GD, 256, 0, stream>>>(act1, wp2, params, act2);
  conv3_kernel<<<BATCH * GD * GD, 256, 0, stream>>>(act2, wp3, params, out);
}
